// PointConv_14078902796568
// MI455X (gfx1250) — matrix operand order repaired and checked by reading, not compiled, not run on hardware
//
#include <hip/hip_runtime.h>
#include <hip/hip_bf16.h>

// ---------------------------------------------------------------------------
// PointConv pipeline for MI455X (gfx1250, wave32, WMMA)
// Shapes: B=16, N=4096, S=1024, K=32, D=Cw=64, OUT=64
// ---------------------------------------------------------------------------

typedef __attribute__((ext_vector_type(16))) _Float16 v16h;
typedef __attribute__((ext_vector_type(8)))  _Float16 v8h;
typedef __attribute__((ext_vector_type(8)))  float    v8f;

#define NB   16
#define NPT  4096
#define NS   1024
#define NK   32
#define NCW  64
#define NDD  64
#define NOUT 64
#define MROWS (NB * NS * NK)        // 524288 MLP rows
#define YROWS (NB * NS)             // 16384 final rows
#define EPSV  1e-5f

// D = A(16x32 f16) x B(32x16 f16) + C(16x16 f32), one instruction, K=32.
__device__ __forceinline__ v8f wmma16x16x32(v16h a, v16h b, v8f c) {
  return __builtin_amdgcn_wmma_f32_16x16x32_f16(
      /*neg_a=*/false, a, /*neg_b=*/false, b,
      /*c_mod=*/(short)0, c, /*reuse_a=*/false, /*reuse_b=*/false);
}

__device__ __forceinline__ v16h cat16(v8h lo, v8h hi) {
  return __builtin_shufflevector(lo, hi, 0, 1, 2, 3, 4, 5, 6, 7,
                                 8, 9, 10, 11, 12, 13, 14, 15);
}

// ---------------------------------------------------------------------------
// 1. Furthest point sampling: one block per batch, sequential 1024 steps.
//    Writes fps order's xyz both to workspace and directly to d_out[0:49152].
// ---------------------------------------------------------------------------
__global__ void __launch_bounds__(512) fps_kernel(
    const float* __restrict__ xyz, int* __restrict__ fpsidx,
    float* __restrict__ nxyz, float* __restrict__ dout) {
  __shared__ float dist[NPT];
  __shared__ float rv[512];
  __shared__ int   ri[512];
  __shared__ float cur[3];
  const int b = blockIdx.x, tid = threadIdx.x;
  const float* X = xyz + (size_t)b * NPT * 3;
  for (int i = tid; i < NPT; i += 512) dist[i] = 1e10f;
  int last = 0;
  for (int s = 0; s < NS; ++s) {
    __syncthreads();
    if (tid == 0) {
      fpsidx[b * NS + s] = last;
      float x0 = X[last * 3 + 0], x1 = X[last * 3 + 1], x2 = X[last * 3 + 2];
      cur[0] = x0; cur[1] = x1; cur[2] = x2;
      int o = (b * NS + s) * 3;
      nxyz[o + 0] = x0; nxyz[o + 1] = x1; nxyz[o + 2] = x2;
      dout[o + 0] = x0; dout[o + 1] = x1; dout[o + 2] = x2;
    }
    __syncthreads();
    float bv = -1.0f; int bi = 0;
    for (int i = tid; i < NPT; i += 512) {
      float dx = X[i * 3 + 0] - cur[0];
      float dy = X[i * 3 + 1] - cur[1];
      float dz = X[i * 3 + 2] - cur[2];
      float d  = dx * dx + dy * dy + dz * dz;
      float nd = fminf(dist[i], d);
      dist[i] = nd;
      if (nd > bv) { bv = nd; bi = i; }
    }
    rv[tid] = bv; ri[tid] = bi;
    __syncthreads();
    for (int off = 256; off > 0; off >>= 1) {
      if (tid < off) {
        if (rv[tid + off] > rv[tid] ||
            (rv[tid + off] == rv[tid] && ri[tid + off] < ri[tid])) {
          rv[tid] = rv[tid + off]; ri[tid] = ri[tid + off];
        }
      }
      __syncthreads();
    }
    last = ri[0];
  }
}

// ---------------------------------------------------------------------------
// 2. kNN: one block per query; 4096 dists in LDS, 32 masked argmin passes.
// ---------------------------------------------------------------------------
__global__ void __launch_bounds__(256) knn_kernel(
    const float* __restrict__ xyz, const float* __restrict__ nxyz,
    int* __restrict__ knn) {
  __shared__ float d[NPT];
  __shared__ float rv[256];
  __shared__ int   ri[256];
  const int bs = blockIdx.x, b = bs >> 10, tid = threadIdx.x;
  const float* X = xyz + (size_t)b * NPT * 3;
  const float q0 = nxyz[bs * 3 + 0], q1 = nxyz[bs * 3 + 1], q2 = nxyz[bs * 3 + 2];
  for (int i = tid; i < NPT; i += 256) {
    float dx = X[i * 3 + 0] - q0, dy = X[i * 3 + 1] - q1, dz = X[i * 3 + 2] - q2;
    d[i] = dx * dx + dy * dy + dz * dz;
  }
  __syncthreads();
  for (int k = 0; k < NK; ++k) {
    float bv = 1e30f; int bi = 0;
    for (int i = tid; i < NPT; i += 256) {
      float v = d[i];
      if (v < bv) { bv = v; bi = i; }
    }
    rv[tid] = bv; ri[tid] = bi;
    __syncthreads();
    for (int off = 128; off > 0; off >>= 1) {
      if (tid < off) {
        if (rv[tid + off] < rv[tid] ||
            (rv[tid + off] == rv[tid] && ri[tid + off] < ri[tid])) {
          rv[tid] = rv[tid + off]; ri[tid] = ri[tid + off];
        }
      }
      __syncthreads();
    }
    if (tid == 0) { knn[(size_t)bs * NK + k] = ri[0]; d[ri[0]] = 1e30f; }
    __syncthreads();
  }
}

// ---------------------------------------------------------------------------
// 3. Precision conversions.
//    feature (B,D,N) f32 -> (B,N,D) f16 (gather-friendly rows).
//    wf (4096,64) f32 -> WMMA-B-fragment-ready f16: [kk][nt][lane][i], so a
//    lane's 16 B-fragment halves are one contiguous 32B block (2x b128 load).
// ---------------------------------------------------------------------------
__global__ void feat_cvt_kernel(const float* __restrict__ f, _Float16* __restrict__ o) {
  int e = blockIdx.x * 256 + threadIdx.x;      // < B*N*D = 4194304
  int b = e >> 18;
  int n = (e >> 6) & (NPT - 1);
  int c = e & (NDD - 1);
  o[e] = (_Float16)f[((size_t)(b * NDD + c)) * NPT + n];
}

__global__ void wf_pack_kernel(const float* __restrict__ f, _Float16* __restrict__ o) {
  int e = blockIdx.x * 256 + threadIdx.x;      // < 4096*64
  int i    = e & 15;
  int lane = (e >> 4) & 31;
  int nt   = (e >> 9) & 3;
  int kk   = e >> 11;
  int half = lane >> 4, n = lane & 15;
  int kg   = kk * 32 + (i & 7) + ((i >> 3) << 4) + half * 8;  // B elem i holds row K = (i&7) + 16*(i>>3) + 8*half
  o[e] = (_Float16)f[kg * 64 + nt * 16 + n];
}

__global__ void zero_stats_kernel(float* __restrict__ p) {
  int i = threadIdx.x;
  if (i < 384) p[i] = 0.0f;
}

// ---------------------------------------------------------------------------
// 4. WeightNet MLP (training-mode BN forces one pass per layer's stats).
//    stage 0/1/2: accumulate sum/sumsq of pre-activation t_stage.
//    stage 3:     write final weights (B,S,K,Cw) as f16.
// ---------------------------------------------------------------------------
__global__ void __launch_bounds__(256) mlp_stage_kernel(
    const float* __restrict__ xyz, const int* __restrict__ knn,
    const float* __restrict__ nxyz,
    const float* __restrict__ w0, const float* __restrict__ b0,
    const float* __restrict__ g0, const float* __restrict__ be0,
    const float* __restrict__ w1, const float* __restrict__ b1,
    const float* __restrict__ g1, const float* __restrict__ be1,
    const float* __restrict__ w2, const float* __restrict__ b2,
    const float* __restrict__ g2, const float* __restrict__ be2,
    float* __restrict__ statraw, const float* __restrict__ statfin,
    _Float16* __restrict__ wout, int stage) {
  __shared__ float sw[2][4096];
  __shared__ float ssum[64], ssq[64];
  const int tid = threadIdx.x;
  if (stage >= 1) for (int i = tid; i < 4096; i += 256) sw[0][i] = w1[i];
  if (stage >= 2) for (int i = tid; i < 4096; i += 256) sw[1][i] = w2[i];
  if (tid < 64) { ssum[tid] = 0.0f; ssq[tid] = 0.0f; }
  __syncthreads();

  const int r = blockIdx.x * 256 + tid;              // < 524288
  const int b = r >> 15, s = (r >> 5) & (NS - 1), n = knn[r];
  float gx[3];
#pragma unroll
  for (int c = 0; c < 3; ++c)
    gx[c] = xyz[((size_t)(b * NPT + n)) * 3 + c] -
            nxyz[((size_t)(b * NS + s)) * 3 + c];

  float h[64];
#pragma unroll
  for (int j = 0; j < 64; ++j)
    h[j] = b0[j] + gx[0] * w0[j] + gx[1] * w0[64 + j] + gx[2] * w0[128 + j];

  const float* gs[3]  = {g0, g1, g2};
  const float* bes[3] = {be0, be1, be2};
  const float* bbs[2] = {b1, b2};
  int done = (stage == 0) ? 0 : -1;
  for (int L = 0; L < 3 && done < 0; ++L) {
    const float* fin = statfin + L * 128;            // mean, invstd pairs
#pragma unroll
    for (int j = 0; j < 64; ++j)
      h[j] = fmaxf(0.0f, gs[L][j] * (h[j] - fin[2 * j]) * fin[2 * j + 1] + bes[L][j]);
    if (L == 2) break;                               // stage==3: weights ready
    float t[64];
    for (int j = 0; j < 64; ++j) {
      float a = bbs[L][j];
      for (int i = 0; i < 64; ++i) a += h[i] * sw[L][i * 64 + j];
      t[j] = a;
    }
#pragma unroll
    for (int j = 0; j < 64; ++j) h[j] = t[j];
    if (stage == L + 1) done = stage;
  }

  if (stage == 3) {
    for (int j = 0; j < 64; ++j) wout[(size_t)r * 64 + j] = (_Float16)h[j];
  } else {
    for (int j = 0; j < 64; ++j) {
      atomicAdd(&ssum[j], h[j]);
      atomicAdd(&ssq[j], h[j] * h[j]);
    }
  }
  __syncthreads();
  if (stage < 3 && tid < 64) {
    atomicAdd(&statraw[stage * 128 + 2 * tid],     ssum[tid]);
    atomicAdd(&statraw[stage * 128 + 2 * tid + 1], ssq[tid]);
  }
}

__global__ void mlp_finalize_kernel(const float* __restrict__ raw,
                                    float* __restrict__ fin, int stage, float cnt) {
  int j = threadIdx.x;                               // 64 threads
  float s = raw[stage * 128 + 2 * j];
  float q = raw[stage * 128 + 2 * j + 1];
  float m = s / cnt;
  float v = q / cnt - m * m;
  fin[stage * 128 + 2 * j]     = m;
  fin[stage * 128 + 2 * j + 1] = rsqrtf(v + EPSV);
}

// ---------------------------------------------------------------------------
// 5. Fused per-centroid einsum + final 1x1 conv.
//    Phase 1: 16x(64x64x32) f16 WMMA, D written into a *fragment-ready*
//             nf panel in LDS: nf element (c,j) lives at
//             [(kk=j>>5)*32 + c + 16*((j>>3)&1)]*16 + (j&7)+((j>>4)&1)*8,
//             i.e. exactly where phase-2's A fragment wants it.
//    Phase 2: nf(16x4096) @ wf(4096x64); A = 2x ds_load_b128 / lane,
//             B = 2x global_load_b128 / lane from the pre-packed wf; K range
//             split across wave pairs, partials combined through LDS.
//    Avoids the 256 MB global nf tensor entirely.
// ---------------------------------------------------------------------------
__global__ void __launch_bounds__(256) fused_pointconv_kernel(
    const _Float16* __restrict__ feat,   // (B,N,D) f16
    const _Float16* __restrict__ wts,    // (B,S,K,Cw) f16
    const _Float16* __restrict__ wfp,    // packed (128,4,32,16) f16
    const int* __restrict__ knn,         // (B,S,K)
    const float* __restrict__ bf,        // (64)
    float* __restrict__ y) {             // (B,S,64) pre-BN
  extern __shared__ char smem[];
  _Float16* nf  = (_Float16*)smem;                             // 128 KB frag panel
  int*      nid = (int*)(smem + 16 * 4096 * sizeof(_Float16)); // [512]

  const int tid  = threadIdx.x;
  const int lane = tid & 31;
  const int wave = tid >> 5;
  const int half = lane >> 4;
  const int mr   = lane & 15;
  const int b    = blockIdx.x >> 6;
  const int s0   = (blockIdx.x & 63) << 4;

  for (int i = tid; i < 512; i += 256)
    nid[i] = knn[((size_t)(b * NS + s0 + (i >> 5))) * NK + (i & 31)];
  __syncthreads();

  // --- Phase 1: nf[c][d][w] = sum_k gf[c][k][d] * wt[c][k][w], K = 32 exact.
  for (int c = wave * 2; c < wave * 2 + 2; ++c) {
    const _Float16* wtc = wts + ((size_t)(b * NS + s0 + c)) * NK * NCW;
    v16h bfr[4];                          // all 4 B tiles, reused over d-tiles
#pragma unroll
    for (int wt_i = 0; wt_i < 4; ++wt_i) {
#pragma unroll
      for (int i = 0; i < 16; ++i)        // B[k][n]: k = i + half*16, n = mr
        bfr[wt_i][i] = wtc[(i + half * 16) * NCW + wt_i * 16 + mr];
    }
    for (int dt = 0; dt < 4; ++dt) {
      v16h a;
#pragma unroll
      for (int i = 0; i < 16; ++i) {      // A[m=d][k] gather, m = mr
        int k = i + half * 16;
        a[i] = feat[((size_t)(b * NPT + nid[c * 32 + k])) * NDD + dt * 16 + mr];
      }
#pragma unroll
      for (int wt_i = 0; wt_i < 4; ++wt_i) {
        v8f acc = {};
        acc = wmma16x16x32(a, bfr[wt_i], acc);
#pragma unroll
        for (int r = 0; r < 8; ++r) {     // D: M = r + half*8, N = mr
          int M  = r + half * 8;
          int j  = (dt * 16 + M) * 64 + wt_i * 16 + mr;
          int kk = j >> 5, jo = j & 31;
          int laneT = c + ((jo >> 3) & 1) * 16;
          int iT    = (jo & 7) + (((jo >> 4) & 1) << 3);
          nf[(kk * 32 + laneT) * 16 + iT] = (_Float16)acc[r];
        }
      }
    }
  }
  __syncthreads();

  // --- Phase 2: y(16x64) = nf(16x4096) @ wf(4096x64) + bf.
  // 8 waves: ntile = wave&3, K-half = wave>>2 (64 steps each).
  const int nt = wave & 3;
  const int kh = wave >> 2;
  v8f acc = {};
  for (int kk = kh * 64; kk < kh * 64 + 64; ++kk) {
    const _Float16* ap = nf + (kk * 32 + lane) * 16;
    const _Float16* bp = wfp + ((size_t)(kk * 4 + nt) * 32 + lane) * 16;
    if (kk + 8 < 128)
      __builtin_prefetch(wfp + ((size_t)((kk + 8) * 4 + nt) * 32 + lane) * 16, 0, 1);
    v16h a  = cat16(*(const v8h*)ap, *(const v8h*)(ap + 8));
    v16h bm = cat16(*(const v8h*)bp, *(const v8h*)(bp + 8));
    acc = wmma16x16x32(a, bm, acc);
  }
  __syncthreads();                        // nf reads done; reuse LDS as scratch
  float* scratch = (float*)smem;          // [4][32][8] floats = 4 KB
  if (wave >= 4) {
#pragma unroll
    for (int r = 0; r < 8; ++r) scratch[((wave - 4) * 32 + lane) * 8 + r] = acc[r];
  }
  __syncthreads();
  if (wave < 4) {
#pragma unroll
    for (int r = 0; r < 8; ++r) {
      float v = acc[r] + scratch[(wave * 32 + lane) * 8 + r];
      int M = r + half * 8;
      y[((size_t)(b * NS + s0 + M)) * NOUT + nt * 16 + mr] = v + bf[nt * 16 + mr];
    }
  }
}

// ---------------------------------------------------------------------------
// 6. Final BN over (B,S) + ReLU + transpose to (B,OUT,S). One block/channel.
// ---------------------------------------------------------------------------
__global__ void __launch_bounds__(256) out_kernel(
    const float* __restrict__ y, const float* __restrict__ gf_,
    const float* __restrict__ bef, float* __restrict__ dout) {
  __shared__ float ss[256], sq[256];
  const int o = blockIdx.x, tid = threadIdx.x;
  float s = 0.0f, q = 0.0f;
  for (int r = tid; r < YROWS; r += 256) {
    float v = y[(size_t)r * NOUT + o];
    s += v; q += v * v;
  }
  ss[tid] = s; sq[tid] = q;
  __syncthreads();
  for (int off = 128; off > 0; off >>= 1) {
    if (tid < off) { ss[tid] += ss[tid + off]; sq[tid] += sq[tid + off]; }
    __syncthreads();
  }
  const float m  = ss[0] / (float)YROWS;
  const float vv = sq[0] / (float)YROWS - m * m;
  const float is = rsqrtf(vv + EPSV);
  const float g  = gf_[o], be = bef[o];
  float* outp = dout + NB * NS * 3;                  // after new_xyz
  for (int r = tid; r < YROWS; r += 256) {
    int b = r >> 10, sidx = r & (NS - 1);
    float val = fmaxf(0.0f, g * (y[(size_t)r * NOUT + o] - m) * is + be);
    outp[((size_t)(b * NOUT + o)) * NS + sidx] = val;
  }
}

// ---------------------------------------------------------------------------
// Workspace layout (bytes, 256-aligned)
// ---------------------------------------------------------------------------
#define OFF_FPSIDX  0u              // 65536    int   (B,S)
#define OFF_KNN     65536u          // 2097152  int   (B,S,K)
#define OFF_NXYZ    2162688u        // 196608   f32   (B,S,3)
#define OFF_FEAT16  2359296u        // 8388608  f16   (B,N,D)
#define OFF_WF16    10747904u       // 524288   f16   packed (128,4,32,16)
#define OFF_WTS16   11272192u       // 67108864 f16   (B,S,K,Cw)
#define OFF_Y       78381056u       // 4194304  f32   (B,S,64)
#define OFF_STATRAW 82575360u       // 1536     f32   [3][64][sum,sumsq]
#define OFF_STATFIN 82576896u       // 1536     f32   [3][64][mean,invstd]

extern "C" void kernel_launch(void* const* d_in, const int* in_sizes, int n_in,
                              void* d_out, int out_size, void* d_ws, size_t ws_size,
                              hipStream_t stream) {
  const float* xyz     = (const float*)d_in[0];
  const float* feature = (const float*)d_in[1];
  const float* w0  = (const float*)d_in[2];
  const float* b0  = (const float*)d_in[3];
  const float* g0  = (const float*)d_in[4];
  const float* be0 = (const float*)d_in[5];
  const float* w1  = (const float*)d_in[6];
  const float* b1  = (const float*)d_in[7];
  const float* g1  = (const float*)d_in[8];
  const float* be1 = (const float*)d_in[9];
  const float* w2  = (const float*)d_in[10];
  const float* b2  = (const float*)d_in[11];
  const float* g2  = (const float*)d_in[12];
  const float* be2 = (const float*)d_in[13];
  const float* wf  = (const float*)d_in[14];
  const float* bf  = (const float*)d_in[15];
  const float* gf  = (const float*)d_in[16];
  const float* bef = (const float*)d_in[17];
  float* out = (float*)d_out;

  char* ws = (char*)d_ws;
  int*       fpsidx  = (int*)(ws + OFF_FPSIDX);
  int*       knn     = (int*)(ws + OFF_KNN);
  float*     nxyz    = (float*)(ws + OFF_NXYZ);
  _Float16*  feat16  = (_Float16*)(ws + OFF_FEAT16);
  _Float16*  wf16    = (_Float16*)(ws + OFF_WF16);
  _Float16*  wts16   = (_Float16*)(ws + OFF_WTS16);
  float*     y       = (float*)(ws + OFF_Y);
  float*     statraw = (float*)(ws + OFF_STATRAW);
  float*     statfin = (float*)(ws + OFF_STATFIN);

  zero_stats_kernel<<<1, 384, 0, stream>>>(statraw);
  fps_kernel<<<NB, 512, 0, stream>>>(xyz, fpsidx, nxyz, out);
  knn_kernel<<<NB * NS, 256, 0, stream>>>(xyz, nxyz, knn);
  feat_cvt_kernel<<<(NB * NPT * NDD) / 256, 256, 0, stream>>>(feature, feat16);
  wf_pack_kernel<<<(4096 * 64) / 256, 256, 0, stream>>>(wf, wf16);

  const int mlp_blocks = MROWS / 256;
  for (int stage = 0; stage < 3; ++stage) {
    mlp_stage_kernel<<<mlp_blocks, 256, 0, stream>>>(
        xyz, knn, nxyz, w0, b0, g0, be0, w1, b1, g1, be1,
        w2, b2, g2, be2, statraw, statfin, wts16, stage);
    mlp_finalize_kernel<<<1, 64, 0, stream>>>(statraw, statfin, stage, (float)MROWS);
  }
  mlp_stage_kernel<<<mlp_blocks, 256, 0, stream>>>(
      xyz, knn, nxyz, w0, b0, g0, be0, w1, b1, g1, be1,
      w2, b2, g2, be2, statraw, statfin, wts16, 3);

  const size_t fused_lds = (size_t)16 * 4096 * sizeof(_Float16) + 512 * sizeof(int);
  fused_pointconv_kernel<<<NB * (NS / 16), 256, fused_lds, stream>>>(
      feat16, wts16, wf16, knn, bf, y);

  out_kernel<<<NOUT, 256, 0, stream>>>(y, gf, bef, out);
}
